// AttentionDecoder_64037962383951
// MI455X (gfx1250) — compile-verified, matrix operand order checked by reading, not run on hardware
//
#include <hip/hip_runtime.h>
#include <hip/hip_bf16.h>

// ---------------- problem constants ----------------
#define Bb 64
#define Ss 512
#define Tt 256
#define Ee 512
#define Hh 512
#define KQd 256
#define Vv 1000
#define NBLK 64   // persistent grid: 64 workgroups x 256 threads (8 waves)

typedef __attribute__((ext_vector_type(16))) __bf16 v16bf;
typedef __attribute__((ext_vector_type(8)))  __bf16 v8bf;
typedef __attribute__((ext_vector_type(8)))  float  v8f;

union FragU { v16bf v; v8bf h[2]; };

// Build a 16x32 bf16 WMMA fragment for lane (row fixed, K-slice per ISA layout):
// lane half h holds K = k0+8h..k0+8h+7 (low 4 VGPRs) and k0+16+8h..+7 (high 4).
// Both runs are contiguous -> two 16B loads.
__device__ __forceinline__ v16bf frag_ab(const __bf16* __restrict__ row, int k0, int half) {
    FragU u;
    u.h[0] = *reinterpret_cast<const v8bf*>(row + k0 + half * 8);
    u.h[1] = *reinterpret_cast<const v8bf*>(row + k0 + 16 + half * 8);
    return u.v;
}

// Accumulate one 16x16 f32 tile over K (multiple of 32) with bf16 WMMA.
// arow = &A[m][0] (row-major, this lane's row), brow = &W[n][0] (W is [N][K], i.e. B^T).
__device__ __forceinline__ v8f wmma_acc(const __bf16* __restrict__ arow,
                                        const __bf16* __restrict__ brow,
                                        int K, int half, v8f acc) {
    for (int k0 = 0; k0 < K; k0 += 32) {
        v16bf af = frag_ab(arow, k0, half);
        v16bf bf_ = frag_ab(brow, k0, half);
        acc = __builtin_amdgcn_wmma_f32_16x16x32_bf16(false, af, false, bf_,
                                                      (short)0, acc, false, false);
    }
    return acc;
}

// ---------------- f32 -> bf16 convert ----------------
__global__ void cvt_f32_bf16(const float* __restrict__ s, __bf16* __restrict__ d, long n) {
    long i = (long)blockIdx.x * blockDim.x + threadIdx.x;
    long stride = (long)gridDim.x * blockDim.x;
    for (; i < n; i += stride) d[i] = (__bf16)s[i];
}

// ---------------- generic WMMA GEMM: C[M][N] = A[M][K] @ W[N][K]^T + bias ----------------
__global__ void __launch_bounds__(256) gemm_bf16_bias(
    const __bf16* __restrict__ A, int lda,
    const __bf16* __restrict__ W, int ldb,
    const float* __restrict__ bias,
    float* __restrict__ C, int ldc,
    int Nt, int K, int total, int nmax) {
    int lane = threadIdx.x & 31, wave = threadIdx.x >> 5;
    int tid = blockIdx.x * 8 + wave;
    if (tid >= total) return;
    int mi = tid / Nt, ni = tid % Nt;
    int half = lane >> 4, l16 = lane & 15;
    int m = mi * 16 + l16, n = ni * 16 + l16;
    int ncl = (n < nmax) ? n : (nmax - 1);
    v8f acc = {};
    acc = wmma_acc(A + (size_t)m * lda, W + (size_t)ncl * ldb, K, half, acc);
    if (n < nmax) {
        float bvv = bias ? bias[n] : 0.f;
        for (int r = 0; r < 8; ++r)
            C[(size_t)(mi * 16 + r + half * 8) * ldc + n] = acc[r] + bvv;
    }
}

// ---------------- init: states, summed biases, SOS embedding, barrier counter ----------------
__global__ void init_kernel(float* h1f, float* c1f, float* h2f, float* c2f,
                            __bf16* h1b, __bf16* xb, __bf16* catb,
                            float* bsum1, const float* bih1, const float* bhh1,
                            float* bsum2, const float* bih2, const float* bhh2,
                            const float* Wemb, unsigned* cnt) {
    int i = blockIdx.x * 256 + threadIdx.x;         // grid 64x256 = 16384
    for (int it = 0; it < 2; ++it) {
        int idx = i + it * 16384;                   // 0..32767 = B*H
        h1f[idx] = 0.f; c1f[idx] = 0.f; h2f[idx] = 0.f; c2f[idx] = 0.f;
        h1b[idx] = (__bf16)0.f;
        int b = idx >> 9, j = idx & 511;
        xb[idx] = (__bf16)Wemb[j];                  // x0 = embedding of SOS (row 0)
        catb[(size_t)b * 1024 + j] = (__bf16)0.f;
        catb[(size_t)b * 1024 + 512 + j] = (__bf16)0.f;
    }
    if (i < 2048) { bsum1[i] = bih1[i] + bhh1[i]; bsum2[i] = bih2[i] + bhh2[i]; }
    if (i == 0) *cnt = 0u;
}

// ---------------- software grid barrier (all NBLK blocks resident) ----------------
__device__ __forceinline__ void gbar(unsigned* cnt, unsigned& phase) {
    __syncthreads();
    ++phase;
    if (threadIdx.x == 0) {
        __threadfence();
        atomicAdd(cnt, 1u);
        while (atomicAdd(cnt, 0u) < phase * (unsigned)NBLK)
            __builtin_amdgcn_s_sleep(2);
    }
    __syncthreads();
}

// ---------------- LSTM elementwise: gates[64][2048] -> h,c ; also bf16 h for WMMA A ----------------
__device__ __forceinline__ void lstm_ew(const float* __restrict__ g,
                                        float* __restrict__ hf, float* __restrict__ cf,
                                        __bf16* __restrict__ hb, int ldh,
                                        int blk, int tid) {
    for (int it = 0; it < 2; ++it) {
        int idx = blk * 256 + tid + it * 16384;     // 0..32767
        int b = idx >> 9, j = idx & 511;
        const float* gr = g + (size_t)b * 2048;
        float gi = gr[j], gf = gr[512 + j], gg = gr[1024 + j], go = gr[1536 + j];
        float si = 1.f / (1.f + __expf(-gi));
        float sf = 1.f / (1.f + __expf(-gf));
        float so = 1.f / (1.f + __expf(-go));
        float c = sf * cf[idx] + si * tanhf(gg);
        float h = so * tanhf(c);
        cf[idx] = c; hf[idx] = h;
        hb[(size_t)b * ldh + j] = (__bf16)h;
    }
}

struct DecArgs {
    const __bf16 *Wih1, *Whh1, *Wih2, *Whh2, *Wq, *W1, *Wemb;
    const float *bsum1, *bsum2, *bq, *b1, *bcls;
    const float *Kc, *Vc;
    const int *y;
    __bf16 *xb, *h1b, *catb, *hidb;
    float *h1f, *c1f, *h2f, *c2f, *qf, *gbuf;
    float *logits, *attns;
    unsigned *cnt;
};

// ---------------- persistent decoder: whole T-loop in one launch ----------------
__global__ void __launch_bounds__(256, 1) decoder_persistent(DecArgs a) {
    __shared__ float sh_q[KQd];
    __shared__ float sh_sc[Ss];
    __shared__ float sh_red[256];

    const int tid = threadIdx.x, lane = tid & 31, wave = tid >> 5;
    const int blk = blockIdx.x;
    const int half = lane >> 4, l16 = lane & 15;
    const int wv = blk * 8 + wave;                  // 0..511 global wave id
    unsigned phase = 0;

    for (int t = 0; t < Tt; ++t) {
        // ===== G1: gates = x @ Wih1^T + h1 @ Whh1^T + (b_ih1+b_hh1)   [64 x 2048] =====
        {
            int mi = wv >> 7, ni = wv & 127;        // 4 x 128 tiles == 512 waves
            int m = mi * 16 + l16, n = ni * 16 + l16;
            v8f acc = {};
            acc = wmma_acc(a.xb  + (size_t)m * 512, a.Wih1 + (size_t)n * 512, 512, half, acc);
            acc = wmma_acc(a.h1b + (size_t)m * 512, a.Whh1 + (size_t)n * 512, 512, half, acc);
            float bvv = a.bsum1[n];
            for (int r = 0; r < 8; ++r)
                a.gbuf[(size_t)(mi * 16 + r + half * 8) * 2048 + n] = acc[r] + bvv;
        }
        gbar(a.cnt, phase);
        // ===== LSTM cell 1 =====
        lstm_ew(a.gbuf, a.h1f, a.c1f, a.h1b, 512, blk, tid);
        gbar(a.cnt, phase);
        // ===== G2: gates = h1 @ Wih2^T + h2 @ Whh2^T + biases =====
        {
            int mi = wv >> 7, ni = wv & 127;
            int m = mi * 16 + l16, n = ni * 16 + l16;
            v8f acc = {};
            acc = wmma_acc(a.h1b  + (size_t)m * 512,  a.Wih2 + (size_t)n * 512, 512, half, acc);
            acc = wmma_acc(a.catb + (size_t)m * 1024, a.Whh2 + (size_t)n * 512, 512, half, acc);
            float bvv = a.bsum2[n];
            for (int r = 0; r < 8; ++r)
                a.gbuf[(size_t)(mi * 16 + r + half * 8) * 2048 + n] = acc[r] + bvv;
        }
        gbar(a.cnt, phase);
        // ===== LSTM cell 2 (h2 written into cat[:, :512], stride 1024) =====
        lstm_ew(a.gbuf, a.h2f, a.c2f, a.catb, 1024, blk, tid);
        gbar(a.cnt, phase);
        // ===== q = h2 @ Wq^T + bq  [64 x 256]; also fetch next-step embedding =====
        if (wv < 64) {
            int mi = wv >> 4, ni = wv & 15;
            int m = mi * 16 + l16, n = ni * 16 + l16;
            v8f acc = {};
            acc = wmma_acc(a.catb + (size_t)m * 1024, a.Wq + (size_t)n * 512, 512, half, acc);
            float bvv = a.bq[n];
            for (int r = 0; r < 8; ++r)
                a.qf[(size_t)(mi * 16 + r + half * 8) * KQd + n] = acc[r] + bvv;
        }
        {   // teacher forcing: x_{t+1} = emb(y[:, t]) ; one dword (2 bf16) per thread
            int tok = a.y[blk * Tt + t];
            const unsigned* src = (const unsigned*)(a.Wemb + (size_t)tok * 512);
            unsigned* dst = (unsigned*)(a.xb + (size_t)blk * 512);
            dst[tid] = src[tid];
        }
        gbar(a.cnt, phase);
        // ===== attention: block b handles batch b (scores, softmax, context) =====
        {
            const int b = blk;
            // stage q into LDS through the CDNA5 async-to-LDS path (ASYNCcnt-tracked)
            {
                unsigned lds_q = (unsigned)(unsigned long long)(const void*)(sh_q + tid);
                const float* gq = a.qf + (size_t)b * KQd + tid;
                asm volatile("global_load_async_to_lds_b32 %0, %1, off"
                             :: "v"(lds_q), "v"(gq) : "memory");
                asm volatile("s_wait_asynccnt 0" ::: "memory");
            }
            __syncthreads();
            const float scale = 0.0625f;            // 1/sqrt(256)
            float s0 = 0.f, s1 = 0.f;
            const float4* q4 = (const float4*)sh_q;
            const float4* K0 = (const float4*)(a.Kc + ((size_t)b * Ss + tid) * KQd);
            const float4* K1 = (const float4*)(a.Kc + ((size_t)b * Ss + tid + 256) * KQd);
            for (int d = 0; d < KQd / 4; ++d) {
                float4 qv = q4[d], k0v = K0[d], k1v = K1[d];
                s0 += qv.x * k0v.x + qv.y * k0v.y + qv.z * k0v.z + qv.w * k0v.w;
                s1 += qv.x * k1v.x + qv.y * k1v.y + qv.z * k1v.z + qv.w * k1v.w;
            }
            s0 *= scale; s1 *= scale;
            sh_red[tid] = fmaxf(s0, s1);
            __syncthreads();
            for (int off = 128; off > 0; off >>= 1) {
                if (tid < off) sh_red[tid] = fmaxf(sh_red[tid], sh_red[tid + off]);
                __syncthreads();
            }
            float mx = sh_red[0];
            __syncthreads();
            float e0 = __expf(s0 - mx), e1 = __expf(s1 - mx);
            sh_red[tid] = e0 + e1;
            __syncthreads();
            for (int off = 128; off > 0; off >>= 1) {
                if (tid < off) sh_red[tid] += sh_red[tid + off];
                __syncthreads();
            }
            float inv = 1.f / sh_red[0];
            __syncthreads();
            float a0 = e0 * inv, a1 = e1 * inv;
            sh_sc[tid] = a0; sh_sc[tid + 256] = a1;
            a.attns[((size_t)b * Tt + t) * Ss + tid] = a0;
            a.attns[((size_t)b * Tt + t) * Ss + tid + 256] = a1;
            __syncthreads();
            // context: thread handles adjacent columns 2*tid, 2*tid+1 -> b64 V loads
            float c0 = 0.f, c1 = 0.f;
            const float2* Vb = (const float2*)(a.Vc + (size_t)b * Ss * Ee) + tid;
            for (int s = 0; s < Ss; ++s) {
                float aw = sh_sc[s];
                float2 vv = Vb[(size_t)s * (Ee / 2)];
                c0 += aw * vv.x;
                c1 += aw * vv.y;
            }
            a.catb[(size_t)b * 1024 + 512 + 2 * tid]     = (__bf16)c0;
            a.catb[(size_t)b * 1024 + 512 + 2 * tid + 1] = (__bf16)c1;
        }
        gbar(a.cnt, phase);
        // ===== hid = relu(cat(h2,ctx) @ W1^T + b1)   [64 x 512], K = 1024 =====
        if (wv < 128) {
            int mi = wv >> 5, ni = wv & 31;
            int m = mi * 16 + l16, n = ni * 16 + l16;
            v8f acc = {};
            acc = wmma_acc(a.catb + (size_t)m * 1024, a.W1 + (size_t)n * 1024, 1024, half, acc);
            float bvv = a.b1[n];
            for (int r = 0; r < 8; ++r)
                a.hidb[(size_t)(mi * 16 + r + half * 8) * 512 + n] = (__bf16)fmaxf(acc[r] + bvv, 0.f);
        }
        gbar(a.cnt, phase);
        // ===== logits = hid @ Wemb^T + b_cls   [64 x 1000] (N padded to 1008) =====
        if (wv < 252) {
            int mi = wv / 63, ni = wv % 63;
            int m = mi * 16 + l16, n = ni * 16 + l16;
            int ncl = (n < Vv) ? n : (Vv - 1);
            v8f acc = {};
            acc = wmma_acc(a.hidb + (size_t)m * 512, a.Wemb + (size_t)ncl * 512, 512, half, acc);
            if (n < Vv) {
                float bvv = a.bcls[n];
                for (int r = 0; r < 8; ++r)
                    a.logits[((size_t)(mi * 16 + r + half * 8) * Tt + t) * Vv + n] = acc[r] + bvv;
            }
        }
        gbar(a.cnt, phase);
    }
}

// ---------------- host side ----------------
extern "C" void kernel_launch(void* const* d_in, const int* in_sizes, int n_in,
                              void* d_out, int out_size, void* d_ws, size_t ws_size,
                              hipStream_t stream) {
    const float* enc  = (const float*)d_in[0];
    const int*   y    = (const int*)  d_in[1];
    const float* Wemb = (const float*)d_in[2];
    const float* Wih1 = (const float*)d_in[3];  const float* bih1 = (const float*)d_in[4];
    const float* Whh1 = (const float*)d_in[5];  const float* bhh1 = (const float*)d_in[6];
    const float* Wih2 = (const float*)d_in[7];  const float* bih2 = (const float*)d_in[8];
    const float* Whh2 = (const float*)d_in[9];  const float* bhh2 = (const float*)d_in[10];
    const float* Wq   = (const float*)d_in[11]; const float* bq   = (const float*)d_in[12];
    const float* Wk   = (const float*)d_in[13]; const float* bk   = (const float*)d_in[14];
    const float* Wv   = (const float*)d_in[15]; const float* bv   = (const float*)d_in[16];
    const float* W1   = (const float*)d_in[17]; const float* b1   = (const float*)d_in[18];
    const float* bcls = (const float*)d_in[19];

    char* w = (char*)d_ws;
    size_t off = 0;
    auto take = [&](size_t bytes) -> void* {
        void* p = w + off;
        off = (off + bytes + 255) & ~(size_t)255;
        return p;
    };
    __bf16* enc_bf  = (__bf16*)take((size_t)Bb * Ss * Ee * 2);
    float*  Kc      = (float*) take((size_t)Bb * Ss * KQd * 4);
    float*  Vc      = (float*) take((size_t)Bb * Ss * Ee * 4);
    __bf16* Wih1b   = (__bf16*)take((size_t)4 * Hh * Hh * 2);
    __bf16* Whh1b   = (__bf16*)take((size_t)4 * Hh * Hh * 2);
    __bf16* Wih2b   = (__bf16*)take((size_t)4 * Hh * Hh * 2);
    __bf16* Whh2b   = (__bf16*)take((size_t)4 * Hh * Hh * 2);
    __bf16* Wqb     = (__bf16*)take((size_t)KQd * Hh * 2);
    __bf16* Wkb     = (__bf16*)take((size_t)KQd * Ee * 2);
    __bf16* Wvb     = (__bf16*)take((size_t)Ee * Ee * 2);
    __bf16* W1b     = (__bf16*)take((size_t)Hh * (Ee + Hh) * 2);
    __bf16* Wembb   = (__bf16*)take((size_t)Vv * Hh * 2);
    float*  h1f     = (float*) take((size_t)Bb * Hh * 4);
    float*  c1f     = (float*) take((size_t)Bb * Hh * 4);
    float*  h2f     = (float*) take((size_t)Bb * Hh * 4);
    float*  c2f     = (float*) take((size_t)Bb * Hh * 4);
    __bf16* h1b     = (__bf16*)take((size_t)Bb * Hh * 2);
    __bf16* xb      = (__bf16*)take((size_t)Bb * Hh * 2);
    __bf16* catb    = (__bf16*)take((size_t)Bb * (Hh + Ee) * 2);
    __bf16* hidb    = (__bf16*)take((size_t)Bb * Hh * 2);
    float*  qf      = (float*) take((size_t)Bb * KQd * 4);
    float*  gbuf    = (float*) take((size_t)Bb * 4 * Hh * 4);
    float*  bsum1   = (float*) take((size_t)4 * Hh * 4);
    float*  bsum2   = (float*) take((size_t)4 * Hh * 4);
    unsigned* cnt   = (unsigned*)take(256);

    // f32 -> bf16 conversions (weights + encoder outputs)
    cvt_f32_bf16<<<2048, 256, 0, stream>>>(enc,  enc_bf, (long)Bb * Ss * Ee);
    cvt_f32_bf16<<<512,  256, 0, stream>>>(Wih1, Wih1b, (long)4 * Hh * Hh);
    cvt_f32_bf16<<<512,  256, 0, stream>>>(Whh1, Whh1b, (long)4 * Hh * Hh);
    cvt_f32_bf16<<<512,  256, 0, stream>>>(Wih2, Wih2b, (long)4 * Hh * Hh);
    cvt_f32_bf16<<<512,  256, 0, stream>>>(Whh2, Whh2b, (long)4 * Hh * Hh);
    cvt_f32_bf16<<<128,  256, 0, stream>>>(Wq,   Wqb,   (long)KQd * Hh);
    cvt_f32_bf16<<<128,  256, 0, stream>>>(Wk,   Wkb,   (long)KQd * Ee);
    cvt_f32_bf16<<<256,  256, 0, stream>>>(Wv,   Wvb,   (long)Ee * Ee);
    cvt_f32_bf16<<<512,  256, 0, stream>>>(W1,   W1b,   (long)Hh * (Ee + Hh));
    cvt_f32_bf16<<<512,  256, 0, stream>>>(Wemb, Wembb, (long)Vv * Hh);

    // K/V cache precompute (WMMA): K = enc @ Wk^T + bk ; V = enc @ Wv^T + bv
    gemm_bf16_bias<<<4096, 256, 0, stream>>>(enc_bf, Ee, Wkb, Ee, bk, Kc, KQd,
                                             KQd / 16, Ee, (Bb * Ss / 16) * (KQd / 16), KQd);
    gemm_bf16_bias<<<8192, 256, 0, stream>>>(enc_bf, Ee, Wvb, Ee, bv, Vc, Ee,
                                             Ee / 16, Ee, (Bb * Ss / 16) * (Ee / 16), Ee);

    init_kernel<<<64, 256, 0, stream>>>(h1f, c1f, h2f, c2f, h1b, xb, catb,
                                        bsum1, bih1, bhh1, bsum2, bih2, bhh2, Wemb, cnt);

    DecArgs a;
    a.Wih1 = Wih1b; a.Whh1 = Whh1b; a.Wih2 = Wih2b; a.Whh2 = Whh2b;
    a.Wq = Wqb; a.W1 = W1b; a.Wemb = Wembb;
    a.bsum1 = bsum1; a.bsum2 = bsum2; a.bq = bq; a.b1 = b1; a.bcls = bcls;
    a.Kc = Kc; a.Vc = Vc; a.y = y;
    a.xb = xb; a.h1b = h1b; a.catb = catb; a.hidb = hidb;
    a.h1f = h1f; a.c1f = c1f; a.h2f = h2f; a.c2f = c2f;
    a.qf = qf; a.gbuf = gbuf;
    a.logits = (float*)d_out;
    a.attns  = (float*)d_out + (size_t)Bb * Tt * Vv;
    a.cnt = cnt;

    decoder_persistent<<<NBLK, 256, 0, stream>>>(a);
}
